// ContrastiveDiffAb_11622181503082
// MI455X (gfx1250) — matrix operand order repaired and checked by reading, not compiled, not run on hardware
//
#include <hip/hip_runtime.h>

// ---------------------------------------------------------------------------
// ContrastiveDiffAb on MI455X (gfx1250):
//   loss = CEdiag(norm(h) @ norm(a)^T) + CEdiag(norm(l) @ norm(a)^T)
// Fused GEMM + streaming logsumexp (cosine sims <= 1 -> no max tracking).
// f16 WMMA with fp32 accumulation; B tiles staged through LDS with
// global_load_async_to_lds_b128 (ASYNCcnt) and shared by all 8 waves;
// each wave keeps a 32x256 A panel in VGPRs; sched_barrier splits the
// B-tile ds_load clause from an uninterrupted 16-WMMA burst.
// ---------------------------------------------------------------------------

typedef __attribute__((ext_vector_type(16))) _Float16 v16h;
typedef __attribute__((ext_vector_type(8)))  _Float16 v8h;
typedef __attribute__((ext_vector_type(8)))  float    v8f;
typedef __attribute__((ext_vector_type(4)))  float    v4f;
typedef __attribute__((__vector_size__(16))) int      v4i;   // matches builtin param type

#define B_ROWS 8192
#define DIMS   256
#define KCH    8                 // 256/32 k-chunks per 16x16 tile
#define JTILES (B_ROWS / 16)     // 512
#define JSPLIT 8                 // j-phases -> 512 workgroups
#define JPW    (JTILES / JSPLIT) // 64 tiles per workgroup
#define BROW   264               // padded LDS row stride (f16) to avoid bank conflicts

#if defined(__has_builtin)
#if __has_builtin(__builtin_amdgcn_global_load_async_to_lds_b128) && \
    __has_builtin(__builtin_amdgcn_s_wait_asynccnt)
#define USE_ASYNC 1
#endif
#if __has_builtin(__builtin_amdgcn_sched_barrier)
#define SCHED_BARRIER() __builtin_amdgcn_sched_barrier(0)
#endif
#endif
#ifndef USE_ASYNC
#define USE_ASYNC 0
#endif
#ifndef SCHED_BARRIER
#define SCHED_BARRIER()
#endif

#if USE_ASYNC
typedef __attribute__((address_space(1))) v4i* as1_v4i_ptr;
typedef __attribute__((address_space(3))) v4i* as3_v4i_ptr;
#endif

union V16HU { v16h v; v8h h[2]; };

// ---------------------------------------------------------------------------
__global__ __launch_bounds__(256) void zero_ws_kernel(float* __restrict__ p, int n) {
  int i = blockIdx.x * 256 + threadIdx.x;
  if (i < n) p[i] = 0.0f;
}

// ---------------------------------------------------------------------------
// Row-normalize fp32 [B,256] -> f16 [B,256]. One wave per row.
// ---------------------------------------------------------------------------
__global__ __launch_bounds__(256) void normalize_f16_kernel(const float* __restrict__ x,
                                                            _Float16* __restrict__ o) {
  const int row  = blockIdx.x * 8 + (threadIdx.x >> 5);
  const int lane = threadIdx.x & 31;
  const float* rp = x + (size_t)row * DIMS + lane * 8;
  v4f a = *(const v4f*)rp;
  v4f b = *(const v4f*)(rp + 4);
  float ss = 0.0f;
#pragma unroll
  for (int i = 0; i < 4; ++i) ss += a[i] * a[i] + b[i] * b[i];
#pragma unroll
  for (int m = 1; m < 32; m <<= 1) ss += __shfl_xor(ss, m, 32);
  const float scale = 1.0f / fmaxf(sqrtf(ss), 1e-8f);   // reference EPS
  v8h out;
#pragma unroll
  for (int i = 0; i < 4; ++i) {
    out[i]     = (_Float16)(a[i] * scale);
    out[i + 4] = (_Float16)(b[i] * scale);
  }
  *(v8h*)(o + (size_t)row * DIMS + lane * 8) = out;
}

// ---------------------------------------------------------------------------
// Stage one 16x256 f16 B tile (8 KB) into a padded LDS buffer.
// 512 B128 chunks, 2 per thread.  Async (ASYNCcnt) path when available.
// ---------------------------------------------------------------------------
__device__ __forceinline__ void stage_tile(const _Float16* __restrict__ a16, int j0,
                                           _Float16* buf, int t) {
#pragma unroll
  for (int c = t; c < 512; c += 256) {
    const int r = c >> 5;             // a-row within tile
    const int o = (c & 31) * 8;       // f16 offset within row
    const _Float16* gp = a16 + (size_t)(j0 + r) * DIMS + o;
    _Float16*       lp = buf + r * BROW + o;
#if USE_ASYNC
    __builtin_amdgcn_global_load_async_to_lds_b128(
        (as1_v4i_ptr)gp, (as3_v4i_ptr)lp, 0, 0);
#else
    *(v8h*)lp = *(const v8h*)gp;      // global_load_b128 + ds_store_b128
#endif
  }
}

__device__ __forceinline__ void write_diag(const v8f& c, int lane, float* __restrict__ dst) {
  // C layout: lane l holds N=l%16, VGPR v holds M=v+8*(l/16)
  // diagonal (M==N): lanes 0..7 (v=lane, M=lane), lanes 24..31 (v=lane-24, M=lane-16)
  if (lane < 8 || lane >= 24) {
    const int dv = (lane < 8) ? lane : (lane - 24);
    float d = 0.0f;
#pragma unroll
    for (int v = 0; v < 8; ++v) d = (dv == v) ? c[v] : d;
    dst[(lane < 8) ? lane : (lane - 16)] = d;
  }
}

__device__ __forceinline__ void reduce_rows(const v8f& se, int m, int khalf,
                                            float* __restrict__ dst) {
#pragma unroll
  for (int v = 0; v < 8; ++v) {
    float s = se[v];
    s += __shfl_xor(s, 1, 32);
    s += __shfl_xor(s, 2, 32);
    s += __shfl_xor(s, 4, 32);
    s += __shfl_xor(s, 8, 32);
    if (m == 0) atomicAdd(dst + khalf * 8 + v, s);
  }
}

// ---------------------------------------------------------------------------
// Main fused kernel.  Workgroup = 128 output rows x 1/8 of the j-range.
// Wave w: matrix = w>>2 (h or l), row-pair = w&3 -> rows i0_0..+15, i0_1..+15.
// All 8 waves sweep the same j-tiles; B tile staged once into LDS (double
// buffered) and reused 8x; each wave reuses it 2x via two accumulators.
// ---------------------------------------------------------------------------
__global__ __launch_bounds__(256) void contrastive_wmma_kernel(
    const _Float16* __restrict__ h16, const _Float16* __restrict__ l16,
    const _Float16* __restrict__ a16,
    float* __restrict__ se_ha, float* __restrict__ se_hl,
    float* __restrict__ dg_ha, float* __restrict__ dg_hl) {
  __shared__ _Float16 Bt[2][16 * BROW];   // 2 x 8.25 KB padded tiles

  const int tid   = threadIdx.x;
  const int lane  = tid & 31;
  const int wave  = tid >> 5;
  const bool isL  = wave >= 4;
  const int  rtp  = wave & 3;             // row-pair index within block
  const int  rb   = blockIdx.x >> 3;      // 128-row block index (0..63)
  const int  jp   = blockIdx.x & 7;       // j phase (0..7)
  const int  jt0  = jp * JPW;
  const int  jtEnd = jt0 + JPW;

  const int d0 = rb * 8 + rtp * 2;        // 16-row tile indices of this wave's rows
  const int d1 = d0 + 1;
  const int i0_0 = d0 * 16;
  const int i0_1 = d1 * 16;

  const _Float16* Abase = isL ? l16 : h16;
  float* sePtr = isL ? se_hl : se_ha;
  float* dgPtr = isL ? dg_hl : dg_ha;

  const int m     = lane & 15;
  const int khalf = lane >> 4;

  // ---- load the two 16x256 A panels into registers (128 VGPRs) ----
  // A 16-bit 16x32 layout: lanes 0-15 K={0..7,16..23}, lanes 16-31 K={8..15,24..31}
  v16h A0[KCH], A1[KCH];
  {
    const _Float16* ar0 = Abase + (size_t)(i0_0 + m) * DIMS;
    const _Float16* ar1 = Abase + (size_t)(i0_1 + m) * DIMS;
#pragma unroll
    for (int k = 0; k < KCH; ++k) {
      V16HU u0, u1;
      u0.h[0] = *(const v8h*)(ar0 + 32 * k +      khalf * 8);
      u0.h[1] = *(const v8h*)(ar0 + 32 * k + 16 + khalf * 8);
      u1.h[0] = *(const v8h*)(ar1 + 32 * k +      khalf * 8);
      u1.h[1] = *(const v8h*)(ar1 + 32 * k + 16 + khalf * 8);
      A0[k] = u0.v;
      A1[k] = u1.v;
    }
  }

  v8f se0 = {0.f,0.f,0.f,0.f,0.f,0.f,0.f,0.f};
  v8f se1 = {0.f,0.f,0.f,0.f,0.f,0.f,0.f,0.f};

  // prologue: stage first tile
  stage_tile(a16, jt0 * 16, Bt[jt0 & 1], tid);
#if USE_ASYNC
  __builtin_amdgcn_s_wait_asynccnt(0);
#endif
  __syncthreads();

  for (int jt = jt0; jt < jtEnd; ++jt) {
    if (jt + 1 < jtEnd)
      stage_tile(a16, (jt + 1) * 16, Bt[(jt + 1) & 1], tid);

    // ---- pull the whole B tile into 64 VGPRs: sched_barrier keeps the
    // ds_load clause separate so all 8 chunk register pairs stay live,
    // then the 16 WMMAs run back-to-back on two independent chains.
    // B 16-bit 32x16 layout: lane = column N, lanes 0-15 K={0..7,16..23}, lanes 16-31 K={8..15,24..31}
    const _Float16* brow = &Bt[jt & 1][m * BROW + khalf * 8];
    V16HU u[KCH];
#pragma unroll
    for (int k = 0; k < KCH; ++k) {
      u[k].h[0] = *(const v8h*)(brow + 32 * k);
      u[k].h[1] = *(const v8h*)(brow + 32 * k + 16);
    }
    SCHED_BARRIER();

    v8f c0 = {0.f,0.f,0.f,0.f,0.f,0.f,0.f,0.f};
    v8f c1 = {0.f,0.f,0.f,0.f,0.f,0.f,0.f,0.f};
#pragma unroll
    for (int k = 0; k < KCH; ++k) {
      c0 = __builtin_amdgcn_wmma_f32_16x16x32_f16(false, A0[k], false, u[k].v,
                                                  (short)0, c0, false, false);
      c1 = __builtin_amdgcn_wmma_f32_16x16x32_f16(false, A1[k], false, u[k].v,
                                                  (short)0, c1, false, false);
    }

    if (jt == d0) write_diag(c0, lane, dgPtr + i0_0);
    if (jt == d1) write_diag(c1, lane, dgPtr + i0_1);

    // streaming sum of exp(sim); sims bounded by 1 -> no max subtraction needed
#pragma unroll
    for (int v = 0; v < 8; ++v) {
      se0[v] += __expf(c0[v]);
      se1[v] += __expf(c1[v]);
    }

#if USE_ASYNC
    __builtin_amdgcn_s_wait_asynccnt(0);
#endif
    __syncthreads();
  }

  // per-row totals: reduce over the 16 column-lanes, combine j-phases via atomics
  reduce_rows(se0, m, khalf, sePtr + i0_0);
  reduce_rows(se1, m, khalf, sePtr + i0_1);
}

// ---------------------------------------------------------------------------
// Final scalar: mean(log(sumexp) - diag) over both similarity matrices.
// ---------------------------------------------------------------------------
__global__ __launch_bounds__(256) void finalize_kernel(const float* __restrict__ se_ha,
                                                       const float* __restrict__ se_hl,
                                                       const float* __restrict__ dg_ha,
                                                       const float* __restrict__ dg_hl,
                                                       float* __restrict__ out) {
  __shared__ float sh[256];
  float s = 0.0f;
  for (int i = threadIdx.x; i < B_ROWS; i += 256)
    s += logf(se_ha[i]) - dg_ha[i] + logf(se_hl[i]) - dg_hl[i];
  sh[threadIdx.x] = s;
  __syncthreads();
  for (int off = 128; off > 0; off >>= 1) {
    if (threadIdx.x < (unsigned)off) sh[threadIdx.x] += sh[threadIdx.x + off];
    __syncthreads();
  }
  if (threadIdx.x == 0) out[0] = sh[0] * (1.0f / (float)B_ROWS);
}

// ---------------------------------------------------------------------------
extern "C" void kernel_launch(void* const* d_in, const int* in_sizes, int n_in,
                              void* d_out, int out_size, void* d_ws, size_t ws_size,
                              hipStream_t stream) {
  const float* hf = (const float*)d_in[0];   // heavy_feat [8192,256] f32
  const float* lf = (const float*)d_in[1];   // light_feat
  const float* af = (const float*)d_in[2];   // antigen_feat

  char* ws = (char*)d_ws;
  const size_t matBytes = (size_t)B_ROWS * DIMS * sizeof(_Float16);  // 4 MiB each
  _Float16* h16 = (_Float16*)(ws);
  _Float16* l16 = (_Float16*)(ws + matBytes);
  _Float16* a16 = (_Float16*)(ws + 2 * matBytes);
  float* se_ha = (float*)(ws + 3 * matBytes);
  float* se_hl = se_ha + B_ROWS;
  float* dg_ha = se_hl + B_ROWS;
  float* dg_hl = dg_ha + B_ROWS;

  zero_ws_kernel<<<(2 * B_ROWS) / 256, 256, 0, stream>>>(se_ha, 2 * B_ROWS);
  normalize_f16_kernel<<<B_ROWS / 8, 256, 0, stream>>>(hf, h16);
  normalize_f16_kernel<<<B_ROWS / 8, 256, 0, stream>>>(lf, l16);
  normalize_f16_kernel<<<B_ROWS / 8, 256, 0, stream>>>(af, a16);
  contrastive_wmma_kernel<<<(B_ROWS / 128) * JSPLIT, 256, 0, stream>>>(
      h16, l16, a16, se_ha, se_hl, dg_ha, dg_hl);
  finalize_kernel<<<1, 256, 0, stream>>>(se_ha, se_hl, dg_ha, dg_hl, (float*)d_out);
}
